// GCN_Estimator_37503654429286
// MI455X (gfx1250) — matrix operand order repaired and checked by reading, not compiled, not run on hardware
//
#include <hip/hip_runtime.h>
#include <hip/hip_bf16.h>

// ---------------- problem constants (match reference) ----------------
#define N_USER   100000
#define M_ITEM   50000
#define N_NODES  (N_USER + M_ITEM)
#define N_EDGES  2400000
#define N_UF     4
#define F_DIM    64
#define E_DIM    32
#define D_EMB    128          // N_UF * E_DIM
#define L_GCN    3
#define N_SAMP   4096

typedef __attribute__((ext_vector_type(16))) _Float16 v16h;
typedef __attribute__((ext_vector_type(8)))  float    v8f;

// ---------------- WMMA fragment helpers (CDNA5 wave32 layouts) ----------------
// A fragment 16x32 f16 from a row-major f32 matrix (rows = M tile, lda in elems).
// lanes 0-15: M=lane,  K = {kb+0..7,  kb+16..23}
// lanes 16-31: M=lane-16, K = {kb+8..15, kb+24..31}
__device__ __forceinline__ v16h load_a_frag_f32(const float* A, int lda, int kb, unsigned lane) {
    const unsigned m    = lane & 15u;
    const unsigned koff = (lane < 16u) ? 0u : 8u;
    const float* row = A + (size_t)m * lda + kb + koff;
    v16h a;
#pragma unroll
    for (int i = 0; i < 8; ++i) {
        a[i]     = (_Float16)row[i];
        a[i + 8] = (_Float16)row[i + 16];
    }
    return a;
}

__device__ __forceinline__ v16h load_a_frag_f16(const _Float16* A, int lda, int kb, unsigned lane) {
    const unsigned m    = lane & 15u;
    const unsigned koff = (lane < 16u) ? 0u : 8u;
    const _Float16* row = A + (size_t)m * lda + kb + koff;
    v16h a;
#pragma unroll
    for (int i = 0; i < 8; ++i) {
        a[i]     = row[i];
        a[i + 8] = row[i + 16];
    }
    return a;
}

// Packed-fragment convention (produced by pack kernels / LDS staging):
//   frag = ((tn * ksteps) + kbIdx); element p = (frag*32 + lane)*16 + i
//   holds B[k][n] with n = tn*16 + (lane&15), k = kbIdx*32 + ((lane<16)?0:8) + (i&7) + ((i>>3)<<4)
// so a B fragment is a single contiguous 32-byte per-lane vector load.

// ---------------- weight pre-pack: f32 (N,K) row-major -> f16 fragments ----------------
__global__ __launch_bounds__(256) void pack_w_kernel(const float* __restrict__ W,
                                                     _Float16* __restrict__ out,
                                                     int K, int tilesN) {
    const int ksteps = K >> 5;
    const int total  = tilesN * ksteps * 512;
    int p = blockIdx.x * blockDim.x + threadIdx.x;
    if (p >= total) return;
    const int i    = p & 15;
    const int lane = (p >> 4) & 31;
    const int frag = p >> 9;
    const int kbIdx = frag % ksteps;
    const int tn    = frag / ksteps;
    const int n = tn * 16 + (lane & 15);
    const int k = kbIdx * 32 + ((lane < 16) ? 0 : 8) + (i & 7) + ((i >> 3) << 4);
    out[p] = (_Float16)W[(size_t)n * K + k];
}

// ---------------- phase A: embedding projection (emb -> acc and h) ----------------
// Weights (Wu & Wi, 32KB as f16 fragments) are staged once per block into LDS in
// packed-fragment order; each wave then reads its B fragment as one 32B ds load.
// smem index(side,f,nt,kbIdx,lane,i) = (((((side*4+f)*2+nt)*2+kbIdx)*32)+lane)*16+i
#define EMB_SMEM_HALFS (2 * N_UF * 2 * 2 * 32 * 16)   // 16384 halfs = 32 KB

__global__ __launch_bounds__(128) void emb_wmma_kernel(
    const float* __restrict__ u_feats, const float* __restrict__ i_feats,
    const float* __restrict__ Wu, const float* __restrict__ Wi,
    float* __restrict__ acc, float* __restrict__ h)
{
    __shared__ __align__(32) _Float16 smem[EMB_SMEM_HALFS];

    // ---- cooperative stage + convert of both weight tensors ----
    for (int p = threadIdx.x; p < EMB_SMEM_HALFS; p += blockDim.x) {
        int r = p;
        const int i    = r & 15;  r >>= 4;
        const int lane = r & 31;  r >>= 5;
        const int kbIdx = r & 1;  r >>= 1;
        const int nt    = r & 1;  r >>= 1;
        const int f     = r & 3;  r >>= 2;
        const int side  = r;                       // 0 = user, 1 = item
        const int k = kbIdx * 32 + ((lane < 16) ? 0 : 8) + (i & 7) + ((i >> 3) << 4);
        const int n = nt * 16 + (lane & 15);
        const float* W = side ? Wi : Wu;
        smem[p] = (_Float16)W[((size_t)f * F_DIM + k) * E_DIM + n];
    }
    __syncthreads();                               // all waves reach this barrier

    const unsigned lane = threadIdx.x & 31u;
    const unsigned wave = threadIdx.x >> 5;
    const unsigned tile = blockIdx.x * (blockDim.x >> 5) + wave;
    const unsigned userTiles = N_USER / 16;        // 6250
    const unsigned itemTiles = M_ITEM / 16;        // 3125
    if (tile >= userTiles + itemTiles) return;     // wave-uniform exit (after barrier)

    const bool isUser = tile < userTiles;
    const int  side   = isUser ? 0 : 1;
    const unsigned lt = isUser ? tile : (tile - userTiles);
    const int nLocal  = lt * 16;
    const int nRows   = isUser ? N_USER : M_ITEM;
    const int gRow0   = isUser ? nLocal : (N_USER + nLocal);
    const float* feats = isUser ? u_feats : i_feats;

    const unsigned n    = lane & 15u;
    const unsigned moff = (lane >> 4) * 8u;

#pragma unroll
    for (int f = 0; f < N_UF; ++f) {
        const float* Af = feats + (size_t)f * nRows * F_DIM + (size_t)nLocal * F_DIM;
#pragma unroll
        for (int nt = 0; nt < 2; ++nt) {           // two 16-wide N halves of E_DIM=32
            v8f c = {};
#pragma unroll
            for (int kbIdx = 0; kbIdx < 2; ++kbIdx) {
                v16h a = load_a_frag_f32(Af, F_DIM, kbIdx * 32, lane);
                const int sidx = (((((side * N_UF + f) * 2 + nt) * 2 + kbIdx) * 32)
                                  + (int)lane) * 16;
                v16h b = *(const v16h*)&smem[sidx];   // 32B ds load
                c = __builtin_amdgcn_wmma_f32_16x16x32_f16(
                        false, a, false, b, (short)0, c, false, false);
            }
            const int colBase = f * E_DIM + nt * 16;
#pragma unroll
            for (int r = 0; r < 8; ++r) {
                const size_t off = (size_t)(gRow0 + moff + r) * D_EMB + colBase + n;
                acc[off] = c[r];
                h[off]   = c[r];
            }
        }
    }
}

// ---------------- phase B: SpMM (scatter) + helpers ----------------
__global__ __launch_bounds__(256) void zero_f4_kernel(float4* __restrict__ p, int n4) {
    int i = blockIdx.x * blockDim.x + threadIdx.x;
    if (i < n4) p[i] = make_float4(0.f, 0.f, 0.f, 0.f);
}

__global__ __launch_bounds__(256) void add_f4_kernel(float4* __restrict__ a,
                                                     const float4* __restrict__ b, int n4) {
    int i = blockIdx.x * blockDim.x + threadIdx.x;
    if (i < n4) {
        float4 x = a[i], y = b[i];
        a[i] = make_float4(x.x + y.x, x.y + y.y, x.z + y.z, x.w + y.w);
    }
}

// one edge per wave32; each lane owns a float4 slice of the 128-dim row
__global__ __launch_bounds__(256) void spmm_kernel(
    const float* __restrict__ vals, const int* __restrict__ rows, const int* __restrict__ cols,
    const float* __restrict__ h, float* __restrict__ hn)
{
    const long long tid = (long long)blockIdx.x * blockDim.x + threadIdx.x;
    const int e = (int)(tid >> 5);
    if (e >= N_EDGES) return;
    const int c4 = ((int)tid & 31) * 4;

    const float v = vals[e];
    const size_t src = (size_t)cols[e] * D_EMB + c4;
    float* dst = hn + (size_t)rows[e] * D_EMB + c4;

    const float4 x = *(const float4*)(h + src);
    __hip_atomic_fetch_add(dst + 0, v * x.x, __ATOMIC_RELAXED, __HIP_MEMORY_SCOPE_AGENT);
    __hip_atomic_fetch_add(dst + 1, v * x.y, __ATOMIC_RELAXED, __HIP_MEMORY_SCOPE_AGENT);
    __hip_atomic_fetch_add(dst + 2, v * x.z, __ATOMIC_RELAXED, __HIP_MEMORY_SCOPE_AGENT);
    __hip_atomic_fetch_add(dst + 3, v * x.w, __ATOMIC_RELAXED, __HIP_MEMORY_SCOPE_AGENT);
}

// ---------------- phase C: MLP head ----------------
// gather user/item GCN rows, apply 1/(L_GCN+1) scale, emit f16 activations
__global__ __launch_bounds__(256) void gather_kernel(
    const float* __restrict__ acc, const int* __restrict__ user_ids,
    const int* __restrict__ item_ids, _Float16* __restrict__ x0)
{
    int t = blockIdx.x * blockDim.x + threadIdx.x;
    if (t >= N_SAMP * 256) return;
    const int s = t >> 8;
    const int d = t & 255;
    float v;
    if (d < D_EMB) v = acc[(size_t)user_ids[s] * D_EMB + d];
    else           v = acc[(size_t)(N_USER + item_ids[s]) * D_EMB + (d - D_EMB)];
    x0[t] = (_Float16)(v * 0.25f);
}

// Y = act(X @ W^T + b); X:(M,K) f16, Bp: pre-packed f16 fragments, Y:(M,N) f16
__global__ __launch_bounds__(128) void fc_wmma_kernel(
    const _Float16* __restrict__ X, const _Float16* __restrict__ Bp,
    const float* __restrict__ bias, _Float16* __restrict__ Y,
    int M, int N, int K, int doRelu)
{
    const unsigned lane = threadIdx.x & 31u;
    const unsigned wave = threadIdx.x >> 5;
    const unsigned tile = blockIdx.x * (blockDim.x >> 5) + wave;
    const unsigned tilesN = (unsigned)N >> 4;
    const unsigned tm = tile / tilesN;
    const unsigned tn = tile % tilesN;
    if ((int)(tm * 16) >= M) return;               // wave-uniform exit

    const int ksteps = K >> 5;
    v8f c = {};
    for (int kbIdx = 0; kbIdx < ksteps; ++kbIdx) {
        v16h a = load_a_frag_f16(X + (size_t)tm * 16 * K, K, kbIdx * 32, lane);
        v16h b = *(const v16h*)(Bp +
                 ((size_t)((tn * ksteps + kbIdx) * 32 + lane)) * 16);  // 32B global load
        c = __builtin_amdgcn_wmma_f32_16x16x32_f16(
                false, a, false, b, (short)0, c, false, false);
    }

    const unsigned n    = lane & 15u;
    const unsigned moff = (lane >> 4) * 8u;
    const float bv = bias[tn * 16 + n];
#pragma unroll
    for (int r = 0; r < 8; ++r) {
        float v = c[r] + bv;
        if (doRelu) v = v > 0.f ? v : 0.f;
        Y[(size_t)(tm * 16 + moff + r) * N + tn * 16 + n] = (_Float16)v;
    }
}

// final 64 -> 1 projection
__global__ __launch_bounds__(256) void out_head_kernel(
    const _Float16* __restrict__ X2, const float* __restrict__ ow,
    const float* __restrict__ ob, float* __restrict__ out)
{
    int s = blockIdx.x * blockDim.x + threadIdx.x;
    if (s >= N_SAMP) return;
    const _Float16* row = X2 + (size_t)s * 64;
    float a = 0.f;
#pragma unroll
    for (int d = 0; d < 64; ++d) a += (float)row[d] * ow[d];
    out[s] = a + ob[0];
}

// ---------------- launcher ----------------
extern "C" void kernel_launch(void* const* d_in, const int* in_sizes, int n_in,
                              void* d_out, int out_size, void* d_ws, size_t ws_size,
                              hipStream_t stream) {
    const float* u_feats  = (const float*)d_in[0];
    const float* i_feats  = (const float*)d_in[1];
    const float* Wu       = (const float*)d_in[2];
    const float* Wi       = (const float*)d_in[3];
    const float* adj_vals = (const float*)d_in[4];
    const float* fc1_w    = (const float*)d_in[5];
    const float* fc1_b    = (const float*)d_in[6];
    const float* fc2_w    = (const float*)d_in[7];
    const float* fc2_b    = (const float*)d_in[8];
    const float* out_w    = (const float*)d_in[9];
    const float* out_b    = (const float*)d_in[10];
    const int*   adj_rows = (const int*)d_in[11];
    const int*   adj_cols = (const int*)d_in[12];
    const int*   user_ids = (const int*)d_in[13];
    const int*   item_ids = (const int*)d_in[14];
    float* out = (float*)d_out;

    // workspace layout (all offsets 32B aligned)
    const size_t nodeBytes = (size_t)N_NODES * D_EMB * sizeof(float);   // 76,800,000
    char* ws   = (char*)d_ws;
    float* acc = (float*)(ws);
    float* h   = (float*)(ws + nodeBytes);
    float* hn  = (float*)(ws + 2 * nodeBytes);
    _Float16* x0   = (_Float16*)(ws + 3 * nodeBytes);
    _Float16* x1   = (_Float16*)((char*)x0 + (size_t)N_SAMP * 256 * sizeof(_Float16));
    _Float16* x2   = (_Float16*)((char*)x1 + (size_t)N_SAMP * 128 * sizeof(_Float16));
    _Float16* pfc1 = (_Float16*)((char*)x2 + (size_t)N_SAMP * 64 * sizeof(_Float16));
    _Float16* pfc2 = (_Float16*)((char*)pfc1 + (size_t)(128 / 16) * (256 / 32) * 512 * sizeof(_Float16));

    // pre-pack fc weights into WMMA fragment order (tiny, runs once up front)
    {
        const int tot1 = (128 / 16) * (256 / 32) * 512;    // 32768
        const int tot2 = (64 / 16) * (128 / 32) * 512;     // 8192
        pack_w_kernel<<<(tot1 + 255) / 256, 256, 0, stream>>>(fc1_w, pfc1, 256, 128 / 16);
        pack_w_kernel<<<(tot2 + 255) / 256, 256, 0, stream>>>(fc2_w, pfc2, 128, 64 / 16);
    }

    // phase A: embeddings -> acc and h
    {
        const int tiles  = N_USER / 16 + M_ITEM / 16;      // 9375
        const int wavesB = 128 / 32;
        const int blocks = (tiles + wavesB - 1) / wavesB;
        emb_wmma_kernel<<<blocks, 128, 0, stream>>>(u_feats, i_feats, Wu, Wi, acc, h);
    }

    // phase B: 3 GCN layers
    const int n4 = N_NODES * D_EMB / 4;                    // 4,800,000
    const int gridF4   = (n4 + 255) / 256;
    const int gridEdge = (int)(((long long)N_EDGES * 32 + 255) / 256);  // 300,000
    float* cur = h; float* nxt = hn;
    for (int l = 0; l < L_GCN; ++l) {
        zero_f4_kernel<<<gridF4, 256, 0, stream>>>((float4*)nxt, n4);
        spmm_kernel<<<gridEdge, 256, 0, stream>>>(adj_vals, adj_rows, adj_cols, cur, nxt);
        add_f4_kernel<<<gridF4, 256, 0, stream>>>((float4*)acc, (const float4*)nxt, n4);
        float* t = cur; cur = nxt; nxt = t;
    }

    // phase C: MLP head
    gather_kernel<<<(N_SAMP * 256) / 256, 256, 0, stream>>>(acc, user_ids, item_ids, x0);
    {
        const int tiles1 = (N_SAMP / 16) * (128 / 16);     // 2048
        fc_wmma_kernel<<<tiles1 / 4, 128, 0, stream>>>(x0, pfc1, fc1_b, x1,
                                                       N_SAMP, 128, 256, 1);
        const int tiles2 = (N_SAMP / 16) * (64 / 16);      // 1024
        fc_wmma_kernel<<<tiles2 / 4, 128, 0, stream>>>(x1, pfc2, fc2_b, x2,
                                                       N_SAMP, 64, 128, 1);
    }
    out_head_kernel<<<(N_SAMP + 255) / 256, 256, 0, stream>>>(x2, out_w, out_b, out);
}
